// Decoder_34445637714134
// MI455X (gfx1250) — matrix operand order repaired and checked by reading, not compiled, not run on hardware
//
#include <hip/hip_runtime.h>

typedef _Float16 h16v __attribute__((ext_vector_type(16)));
typedef _Float16 h8v  __attribute__((ext_vector_type(8)));
typedef float    f8v  __attribute__((ext_vector_type(8)));

#define DMODEL 512
#define NHEAD  8
#define DHEAD  64
#define BB     16
#define TDQ    256
#define TEK    1024
#define NL     6
#define DFFN   2048

__device__ __forceinline__ h16v cat8(h8v lo, h8v hi) {
  return __builtin_shufflevector(lo, hi, 0,1,2,3,4,5,6,7,8,9,10,11,12,13,14,15);
}
__device__ __forceinline__ f8v wmma16(h16v a, h16v b, f8v c) {
  return __builtin_amdgcn_wmma_f32_16x16x32_f16(false, a, false, b, (short)0, c, false, false);
}

// ---------------------------------------------------------------------------
// Weight transpose + f32->f16:  Wt[n*K + k] = (f16) W[k*N + n]
// ---------------------------------------------------------------------------
__global__ void wt_kernel(const float* __restrict__ W, _Float16* __restrict__ Wt,
                          int K, int N) {
  long idx = (long)blockIdx.x * blockDim.x + threadIdx.x;
  long tot = (long)K * N;
  if (idx >= tot) return;
  int n = (int)(idx / K);
  int k = (int)(idx % K);
  Wt[idx] = (_Float16)W[(size_t)k * N + n];
}

// ---------------------------------------------------------------------------
// f32 -> f16 flat convert (encoder outputs)
// ---------------------------------------------------------------------------
__global__ void cvt_kernel(const float* __restrict__ s, _Float16* __restrict__ d, long n) {
  long i = (long)blockIdx.x * blockDim.x + threadIdx.x;
  if (i < n) d[i] = (_Float16)s[i];
}

// ---------------------------------------------------------------------------
// Embedding * sqrt(D) + sinusoidal PE -> f16 [4096,512]
// ---------------------------------------------------------------------------
__global__ void embed_kernel(const int* __restrict__ tgt, const float* __restrict__ emb,
                             _Float16* __restrict__ out) {
  int idx = blockIdx.x * blockDim.x + threadIdx.x;   // 0 .. 4096*512-1
  int d = idx & (DMODEL - 1);
  int row = idx >> 9;
  int t = row & (TDQ - 1);
  int tok = tgt[row];
  float div = __expf((float)(d & ~1) * (-9.210340371976184f / (float)DMODEL));
  float pe = (d & 1) ? cosf((float)t * div) : sinf((float)t * div);
  out[idx] = (_Float16)(emb[(size_t)tok * DMODEL + d] * 22.627416997969522f + pe);
}

// ---------------------------------------------------------------------------
// LayerNorm: f32 x[row] -> f16 out[row], one wave per row (512 elems)
// ---------------------------------------------------------------------------
__global__ __launch_bounds__(256) void ln_kernel(const float* __restrict__ x,
                                                 const float* __restrict__ g,
                                                 const float* __restrict__ bt,
                                                 _Float16* __restrict__ out, int M) {
  int lane = threadIdx.x & 31, wave = threadIdx.x >> 5;
  int row = blockIdx.x * 8 + wave;
  if (row >= M) return;
  const float* xr = x + (size_t)row * DMODEL;
  float v[16];
  float s = 0.f;
#pragma unroll
  for (int i = 0; i < 16; ++i) { v[i] = xr[lane + 32 * i]; s += v[i]; }
#pragma unroll
  for (int m = 1; m < 32; m <<= 1) s += __shfl_xor(s, m, 32);
  float mean = s * (1.f / DMODEL);
  float vs = 0.f;
#pragma unroll
  for (int i = 0; i < 16; ++i) { float d = v[i] - mean; vs += d * d; }
#pragma unroll
  for (int m = 1; m < 32; m <<= 1) vs += __shfl_xor(vs, m, 32);
  float rstd = rsqrtf(vs * (1.f / DMODEL) + 1e-5f);
  _Float16* orow = out + (size_t)row * DMODEL;
#pragma unroll
  for (int i = 0; i < 16; ++i) {
    int c = lane + 32 * i;
    orow[c] = (_Float16)((v[i] - mean) * rstd * g[c] + bt[c]);
  }
}

// ---------------------------------------------------------------------------
// V repack: (b*T+t, 512) row-major -> Vt[((b*H+h)*64+d)*T + t]
// ---------------------------------------------------------------------------
__global__ void vswap_kernel(const _Float16* __restrict__ Vr, _Float16* __restrict__ Vt,
                             int T) {
  long idx = (long)blockIdx.x * blockDim.x + threadIdx.x;
  long tot = (long)BB * T * DMODEL;
  if (idx >= tot) return;
  int t = (int)(idx % T);
  long r = idx / T;
  int d = (int)(r % DHEAD);
  long r2 = r / DHEAD;
  int h = (int)(r2 % NHEAD);
  int b = (int)(r2 / NHEAD);
  Vt[idx] = Vr[((size_t)(b * T + t)) * DMODEL + h * DHEAD + d];
}

// ---------------------------------------------------------------------------
// GEMM fragment loaders (k must be a compile-time constant at the call sites
// inside the unrolled chunk so loads use immediate offsets)
// ---------------------------------------------------------------------------
__device__ __forceinline__ void ldA2(const _Float16* p0, const _Float16* p1, int k,
                                     h16v a[2]) {
  a[0] = cat8(*(const h8v*)(p0 + k), *(const h8v*)(p0 + k + 8));
  a[1] = cat8(*(const h8v*)(p1 + k), *(const h8v*)(p1 + k + 8));
}
__device__ __forceinline__ void ldB4(const _Float16* p0, const _Float16* p1,
                                     const _Float16* p2, const _Float16* p3, int k,
                                     h16v b[4]) {
  b[0] = cat8(*(const h8v*)(p0 + k), *(const h8v*)(p0 + k + 8));
  b[1] = cat8(*(const h8v*)(p1 + k), *(const h8v*)(p1 + k + 8));
  b[2] = cat8(*(const h8v*)(p2 + k), *(const h8v*)(p2 + k + 8));
  b[3] = cat8(*(const h8v*)(p3 + k), *(const h8v*)(p3 + k + 8));
}
__device__ __forceinline__ void mac8(const h16v a[2], const h16v b[4], f8v c[8]) {
#pragma unroll
  for (int t = 0; t < 4; ++t) {
    c[t]     = wmma16(a[0], b[t], c[t]);
    c[4 + t] = wmma16(a[1], b[t], c[4 + t]);
  }
}

// ---------------------------------------------------------------------------
// GEMM: C[M,N] = A[M,K](f16 rm) @ Wt[N,K](f16 rm, pre-transposed B) + bias
//       optional residual add (f32), optional ReLU; f32 and/or f16 output.
// 1 wave -> 32x64 tile (8 WMMA / 32-K step).  K processed in 256-element
// chunks: inner loop fully unrolled with compile-time immediate offsets and
// ping-pong register sets; 6 pointer bumps per 64 WMMAs.  Cross-chunk
// prefetch guarded by a wave-uniform branch (EXEC stays all-ones).
// block = 8 waves -> 32x512 strip.  K must be a multiple of 256 (512/2048).
// ---------------------------------------------------------------------------
__global__ __launch_bounds__(256) void gemm_kernel(
    const _Float16* __restrict__ A, const _Float16* __restrict__ Bt,
    const float* __restrict__ bias, const float* __restrict__ res,
    float* __restrict__ oF, _Float16* __restrict__ oH,
    int M, int N, int K, int relu) {
  int lane = threadIdx.x & 31, wave = threadIdx.x >> 5;
  int m0 = blockIdx.x * 32;
  int n0 = blockIdx.y * 512 + wave * 64;
  if (n0 >= N) return;
  int mr = lane & 15, hh = lane >> 4;

  const _Float16* pa0 = A + (size_t)(m0 + mr) * K + hh * 16;
  const _Float16* pa1 = A + (size_t)(m0 + 16 + mr) * K + hh * 16;
  const _Float16* pb0 = Bt + (size_t)(n0 + mr) * K + hh * 16;
  const _Float16* pb1 = Bt + (size_t)(n0 + 16 + mr) * K + hh * 16;
  const _Float16* pb2 = Bt + (size_t)(n0 + 32 + mr) * K + hh * 16;
  const _Float16* pb3 = Bt + (size_t)(n0 + 48 + mr) * K + hh * 16;

  f8v cc[8] = {};
  h16v a0[2], b0[4], a1[2], b1[4];
  ldA2(pa0, pa1, 0, a0);
  ldB4(pb0, pb1, pb2, pb3, 0, b0);

  for (int ch = 0; ch < K; ch += 256) {
#pragma unroll
    for (int kk = 0; kk < 256; kk += 64) {
      // prefetch set 1 at kk+32 (immediate offsets), MAC on set 0
      ldA2(pa0, pa1, kk + 32, a1);
      ldB4(pb0, pb1, pb2, pb3, kk + 32, b1);
      mac8(a0, b0, cc);
      // prefetch set 0 at kk+64 for all but the chunk's last step
      if (kk + 64 < 256) {
        ldA2(pa0, pa1, kk + 64, a0);
        ldB4(pb0, pb1, pb2, pb3, kk + 64, b0);
      }
      mac8(a1, b1, cc);
    }
    pa0 += 256; pa1 += 256;
    pb0 += 256; pb1 += 256; pb2 += 256; pb3 += 256;
    if (ch + 256 < K) {   // wave-uniform: prologue loads for next chunk
      ldA2(pa0, pa1, 0, a0);
      ldB4(pb0, pb1, pb2, pb3, 0, b0);
    }
  }
#pragma unroll
  for (int u = 0; u < 2; ++u) {
#pragma unroll
    for (int t = 0; t < 4; ++t) {
#pragma unroll
      for (int r = 0; r < 8; ++r) {
        int row = m0 + u * 16 + r + 8 * hh;
        int col = n0 + t * 16 + mr;
        float v = cc[4 * u + t][r] + bias[col];
        if (res) v += res[(size_t)row * N + col];
        if (relu) v = v < 0.f ? 0.f : v;
        if (oF) oF[(size_t)row * N + col] = v;
        if (oH) oH[(size_t)row * N + col] = (_Float16)v;
      }
    }
  }
}

// ---------------------------------------------------------------------------
// Flash attention: 1 wave per (b, h, 16-query tile). 32 keys / iteration.
// Q: (b*Tq+t, 512) f16  K: (b*Tk+t, 512) f16  Vt: (b,h,dh,Tk) f16
// ctx out: (b*Tq+t, 512) f16. lens masks keys; causal adds key>query mask.
// K and V fragments are loaded up-front each iteration so they overlap the
// VALU-heavy online-softmax block and the LDS re-fragmentation wait.
// ---------------------------------------------------------------------------
__global__ __launch_bounds__(256) void attn_kernel(
    const _Float16* __restrict__ Q, const _Float16* __restrict__ Kc,
    const _Float16* __restrict__ Vt, const int* __restrict__ lens,
    _Float16* __restrict__ ctx, int Tq, int Tk, int causal) {
  __shared__ __align__(16) _Float16 plds[8][16 * 32];
  int lane = threadIdx.x & 31, wave = threadIdx.x >> 5;
  int wid = blockIdx.x * 8 + wave;
  int qtiles = Tq / 16;
  int qt = wid % qtiles;
  int h = (wid / qtiles) % NHEAD;
  int b = wid / (qtiles * NHEAD);
  int mr = lane & 15, hh = lane >> 4;
  int klen = lens[b];

  const _Float16* qbase = Q + (size_t)(b * Tq + qt * 16 + mr) * DMODEL + h * DHEAD;
  h16v qa[2];
#pragma unroll
  for (int cth = 0; cth < 2; ++cth) {
    const _Float16* qp = qbase + cth * 32 + hh * 16;
    qa[cth] = cat8(*(const h8v*)qp, *(const h8v*)(qp + 8));
  }

  f8v acc[4] = {};
  float mrun[8], lrun[8];
#pragma unroll
  for (int r = 0; r < 8; ++r) { mrun[r] = -1e30f; lrun[r] = 0.f; }

  for (int kb = 0; kb < Tk; kb += 32) {
    // issue all K and V fragment loads for this step first
    h16v kf[2][2], vf[4];
#pragma unroll
    for (int j = 0; j < 2; ++j) {
#pragma unroll
      for (int cth = 0; cth < 2; ++cth) {
        const _Float16* kp = Kc + (size_t)(b * Tk + kb + j * 16 + mr) * DMODEL +
                             h * DHEAD + cth * 32 + hh * 16;
        kf[j][cth] = cat8(*(const h8v*)kp, *(const h8v*)(kp + 8));
      }
    }
#pragma unroll
    for (int t = 0; t < 4; ++t) {
      const _Float16* vp = Vt + (size_t)((b * NHEAD + h) * DHEAD + t * 16 + mr) * Tk +
                           kb + hh * 16;
      vf[t] = cat8(*(const h8v*)vp, *(const h8v*)(vp + 8));
    }
    // scores: 16 queries x 32 keys (two 16x16 n-tiles), K-dim = dh = 64
    f8v s[2] = {};
#pragma unroll
    for (int j = 0; j < 2; ++j)
#pragma unroll
      for (int cth = 0; cth < 2; ++cth)
        s[j] = wmma16(qa[cth], kf[j][cth], s[j]);
    // scale + mask
#pragma unroll
    for (int r = 0; r < 8; ++r) {
      int qrow = qt * 16 + r + 8 * hh;
#pragma unroll
      for (int j = 0; j < 2; ++j) {
        int col = kb + j * 16 + mr;
        float v = s[j][r] * 0.125f;
        bool msk = (col >= klen) || (causal && (col > qrow));
        s[j][r] = msk ? -1e9f : v;
      }
    }
    // online softmax (rows live across 16-lane groups of the C layout)
#pragma unroll
    for (int r = 0; r < 8; ++r) {
      float mx = fmaxf(s[0][r], s[1][r]);
#pragma unroll
      for (int m = 1; m < 16; m <<= 1) mx = fmaxf(mx, __shfl_xor(mx, m, 32));
      float mnew = fmaxf(mrun[r], mx);
      float alpha = __expf(mrun[r] - mnew);
      float p0 = __expf(s[0][r] - mnew);
      float p1 = __expf(s[1][r] - mnew);
      float rs = p0 + p1;
#pragma unroll
      for (int m = 1; m < 16; m <<= 1) rs += __shfl_xor(rs, m, 32);
      lrun[r] = lrun[r] * alpha + rs;
      mrun[r] = mnew;
      s[0][r] = p0; s[1][r] = p1;
#pragma unroll
      for (int t = 0; t < 4; ++t) acc[t][r] *= alpha;
    }
    // re-fragment P (C-layout -> A-fragment) via LDS, same wave
    _Float16* pl = plds[wave];
#pragma unroll
    for (int r = 0; r < 8; ++r) {
#pragma unroll
      for (int j = 0; j < 2; ++j)
        pl[(r + 8 * hh) * 32 + j * 16 + mr] = (_Float16)s[j][r];
    }
    asm volatile("s_wait_dscnt 0" ::: "memory");
    const _Float16* prow = pl + mr * 32 + hh * 16;
    h16v pf = cat8(*(const h8v*)prow, *(const h8v*)(prow + 8));
    // acc(16x64) += P(16x32) @ V(32x64)
#pragma unroll
    for (int t = 0; t < 4; ++t) acc[t] = wmma16(pf, vf[t], acc[t]);
  }
  // normalize + store ctx (f16, per-head 64 cols)
#pragma unroll
  for (int t = 0; t < 4; ++t) {
#pragma unroll
    for (int r = 0; r < 8; ++r) {
      float o = acc[t][r] / lrun[r];
      ctx[(size_t)(b * Tq + qt * 16 + r + 8 * hh) * DMODEL + h * DHEAD + t * 16 + mr] =
          (_Float16)o;
    }
  }
}

// ---------------------------------------------------------------------------
// Host side
// ---------------------------------------------------------------------------
extern "C" void kernel_launch(void* const* d_in, const int* in_sizes, int n_in,
                              void* d_out, int out_size, void* d_ws, size_t ws_size,
                              hipStream_t stream) {
  (void)in_sizes; (void)n_in; (void)out_size; (void)ws_size;
  const int*   targets = (const int*)d_in[0];
  const int*   tlens   = (const int*)d_in[1];
  const float* enc     = (const float*)d_in[2];
  const int*   elens   = (const int*)d_in[3];
  const float* emb     = (const float*)d_in[4];
  const float* W_in    = (const float*)d_in[5];
  const float* b_in    = (const float*)d_in[6];
  const float* ln1g = (const float*)d_in[7],  *ln1b = (const float*)d_in[8];
  const float* Wq1  = (const float*)d_in[9],  *bq1  = (const float*)d_in[10];
  const float* Wk1  = (const float*)d_in[11], *bk1  = (const float*)d_in[12];
  const float* Wv1  = (const float*)d_in[13], *bv1  = (const float*)d_in[14];
  const float* Wo1  = (const float*)d_in[15], *bo1  = (const float*)d_in[16];
  const float* ln2g = (const float*)d_in[17], *ln2b = (const float*)d_in[18];
  const float* Wq2  = (const float*)d_in[19], *bq2  = (const float*)d_in[20];
  const float* Wk2  = (const float*)d_in[21], *bk2  = (const float*)d_in[22];
  const float* Wv2  = (const float*)d_in[23], *bv2  = (const float*)d_in[24];
  const float* Wo2  = (const float*)d_in[25], *bo2  = (const float*)d_in[26];
  const float* ln3g = (const float*)d_in[27], *ln3b = (const float*)d_in[28];
  const float* Wff1 = (const float*)d_in[29], *bff1 = (const float*)d_in[30];
  const float* Wff2 = (const float*)d_in[31], *bff2 = (const float*)d_in[32];

  char* cur = (char*)d_ws;
  auto take = [&](size_t bytes) -> void* {
    void* p = (void*)cur;
    cur += (bytes + 255) & ~(size_t)255;
    return p;
  };
  const size_t DD = (size_t)DMODEL * DMODEL;          // 512*512
  const size_t DF = (size_t)DMODEL * DFFN;            // 512*2048

  _Float16* WinT = (_Float16*)take(DD * 2);
  _Float16 *q1T[NL], *k1T[NL], *v1T[NL], *o1T[NL];
  _Float16 *q2T[NL], *k2T[NL], *v2T[NL], *o2T[NL];
  _Float16 *f1T[NL], *f2T[NL];
  for (int l = 0; l < NL; ++l) {
    q1T[l] = (_Float16*)take(DD * 2); k1T[l] = (_Float16*)take(DD * 2);
    v1T[l] = (_Float16*)take(DD * 2); o1T[l] = (_Float16*)take(DD * 2);
    q2T[l] = (_Float16*)take(DD * 2); k2T[l] = (_Float16*)take(DD * 2);
    v2T[l] = (_Float16*)take(DD * 2); o2T[l] = (_Float16*)take(DD * 2);
    f1T[l] = (_Float16*)take(DF * 2); f2T[l] = (_Float16*)take(DF * 2);
  }
  const int MDEC = BB * TDQ;     // 4096 decoder rows
  const int MENC = BB * TEK;     // 16384 encoder rows
  float*    xbuf = (float*)take((size_t)MDEC * DMODEL * 4);
  _Float16* hbuf = (_Float16*)take((size_t)MDEC * DMODEL * 2);
  _Float16* qbuf = (_Float16*)take((size_t)MDEC * DMODEL * 2);
  _Float16* kbuf = (_Float16*)take((size_t)MENC * DMODEL * 2);
  _Float16* vrow = (_Float16*)take((size_t)MENC * DMODEL * 2);
  _Float16* vtb  = (_Float16*)take((size_t)MENC * DMODEL * 2);
  _Float16* ctxb = (_Float16*)take((size_t)MDEC * DMODEL * 2);
  _Float16* ffb  = (_Float16*)take((size_t)MDEC * DFFN * 2);
  _Float16* encb = (_Float16*)take((size_t)MENC * DMODEL * 2);

  auto wt = [&](const float* W, _Float16* dst, int K, int N) {
    long tot = (long)K * N;
    wt_kernel<<<dim3((unsigned)((tot + 255) / 256)), 256, 0, stream>>>(W, dst, K, N);
  };
  auto gemm = [&](const _Float16* A, const _Float16* Bt, const float* bias,
                  const float* res, float* oF, _Float16* oH,
                  int M, int N, int K, int relu) {
    gemm_kernel<<<dim3(M / 32, N / 512), 256, 0, stream>>>(A, Bt, bias, res, oF, oH,
                                                           M, N, K, relu);
  };

  // --- pack all weights (f32 -> transposed f16) ---
  wt(W_in, WinT, DMODEL, DMODEL);
  for (int l = 0; l < NL; ++l) {
    wt(Wq1 + l * DD, q1T[l], DMODEL, DMODEL);
    wt(Wk1 + l * DD, k1T[l], DMODEL, DMODEL);
    wt(Wv1 + l * DD, v1T[l], DMODEL, DMODEL);
    wt(Wo1 + l * DD, o1T[l], DMODEL, DMODEL);
    wt(Wq2 + l * DD, q2T[l], DMODEL, DMODEL);
    wt(Wk2 + l * DD, k2T[l], DMODEL, DMODEL);
    wt(Wv2 + l * DD, v2T[l], DMODEL, DMODEL);
    wt(Wo2 + l * DD, o2T[l], DMODEL, DMODEL);
    wt(Wff1 + l * DF, f1T[l], DMODEL, DFFN);
    wt(Wff2 + l * DF, f2T[l], DFFN, DMODEL);
  }
  long encN = (long)MENC * DMODEL;
  cvt_kernel<<<dim3((unsigned)((encN + 255) / 256)), 256, 0, stream>>>(enc, encb, encN);

  // --- embedding + PE -> input projection (residual stream xbuf, f32) ---
  embed_kernel<<<dim3(MDEC * DMODEL / 256), 256, 0, stream>>>(targets, emb, hbuf);
  gemm(hbuf, WinT, b_in, nullptr, xbuf, nullptr, MDEC, DMODEL, DMODEL, 0);

  const int ATTN_BLOCKS = BB * NHEAD * (TDQ / 16) / 8;   // 256
  for (int l = 0; l < NL; ++l) {
    // self-attention
    ln_kernel<<<dim3(MDEC / 8), 256, 0, stream>>>(xbuf, ln1g + l * DMODEL,
                                                  ln1b + l * DMODEL, hbuf, MDEC);
    gemm(hbuf, q1T[l], bq1 + l * DMODEL, nullptr, nullptr, qbuf, MDEC, DMODEL, DMODEL, 0);
    gemm(hbuf, k1T[l], bk1 + l * DMODEL, nullptr, nullptr, kbuf, MDEC, DMODEL, DMODEL, 0);
    gemm(hbuf, v1T[l], bv1 + l * DMODEL, nullptr, nullptr, vrow, MDEC, DMODEL, DMODEL, 0);
    vswap_kernel<<<dim3((unsigned)(((long)MDEC * DMODEL + 255) / 256)), 256, 0, stream>>>(
        vrow, vtb, TDQ);
    attn_kernel<<<dim3(ATTN_BLOCKS), 256, 0, stream>>>(qbuf, kbuf, vtb, tlens, ctxb,
                                                       TDQ, TDQ, 1);
    gemm(ctxb, o1T[l], bo1 + l * DMODEL, xbuf, xbuf, nullptr, MDEC, DMODEL, DMODEL, 0);

    // cross-attention
    ln_kernel<<<dim3(MDEC / 8), 256, 0, stream>>>(xbuf, ln2g + l * DMODEL,
                                                  ln2b + l * DMODEL, hbuf, MDEC);
    gemm(hbuf, q2T[l], bq2 + l * DMODEL, nullptr, nullptr, qbuf, MDEC, DMODEL, DMODEL, 0);
    gemm(encb, k2T[l], bk2 + l * DMODEL, nullptr, nullptr, kbuf, MENC, DMODEL, DMODEL, 0);
    gemm(encb, v2T[l], bv2 + l * DMODEL, nullptr, nullptr, vrow, MENC, DMODEL, DMODEL, 0);
    vswap_kernel<<<dim3((unsigned)(((long)MENC * DMODEL + 255) / 256)), 256, 0, stream>>>(
        vrow, vtb, TEK);
    attn_kernel<<<dim3(ATTN_BLOCKS), 256, 0, stream>>>(qbuf, kbuf, vtb, elens, ctxb,
                                                       TDQ, TEK, 0);
    gemm(ctxb, o2T[l], bo2 + l * DMODEL, xbuf, xbuf, nullptr, MDEC, DMODEL, DMODEL, 0);

    // FFN
    ln_kernel<<<dim3(MDEC / 8), 256, 0, stream>>>(xbuf, ln3g + l * DMODEL,
                                                  ln3b + l * DMODEL, hbuf, MDEC);
    gemm(hbuf, f1T[l], bff1 + l * DFFN, nullptr, nullptr, ffb, MDEC, DFFN, DMODEL, 1);
    float* dst = (l == NL - 1) ? (float*)d_out : xbuf;
    gemm(ffb, f2T[l], bff2 + l * DMODEL, xbuf, dst, nullptr, MDEC, DMODEL, DFFN, 0);
  }
}
